// CrossAttention_62758062129707
// MI455X (gfx1250) — matrix operand order repaired and checked by reading, not compiled, not run on hardware
//
#include <hip/hip_runtime.h>
#include <hip/hip_bf16.h>

// ---------------- problem constants ----------------
#define BATCH      65536
#define DMODEL     1024      // QUERY_DIM == CROSS_DIM == INNER
#define HEADS      16
#define DIM_HEAD   64
#define SCALE      0.125f    // 64^-0.5
#define ROWS       32        // batch rows per workgroup (2 row-tiles of 16)
#define ROWTILES   2
#define NTHREADS   512       // 16 wave32 waves -> 4 waves/SIMD for latency hiding
#define NWAVES     16
#define COLTILES   (DMODEL / 16)           // 64
#define TILES_PER_WAVE (COLTILES / NWAVES) // 4 -> only 64 acc VGPRs/wave
#define ROWS_PER_WAVE  (ROWS / NWAVES)     // 2 (attention rows per wave)

typedef __bf16 bf16;
typedef __attribute__((ext_vector_type(16))) __bf16 v16bf;
typedef __attribute__((ext_vector_type(8)))  __bf16 bf16x8;
typedef __attribute__((ext_vector_type(8)))  float  v8f;

// ---------------- helpers ----------------
static __device__ inline bf16x8 zero8() {
    bf16x8 z;
#pragma unroll
    for (int i = 0; i < 8; ++i) z[i] = (bf16)0.0f;
    return z;
}
static __device__ inline v16bf zero16() {
    v16bf z;
#pragma unroll
    for (int i = 0; i < 16; ++i) z[i] = (bf16)0.0f;
    return z;
}
static __device__ inline v8f zerof8() {
    v8f z = {0.f, 0.f, 0.f, 0.f, 0.f, 0.f, 0.f, 0.f};
    return z;
}

// A-operand fragment (16x32 bf16, M x K): lane = row (lane&15),
// low lanes hold K = {k0..k0+7, k0+16..k0+23}, high lanes the +8 halves.
template <typename P>
static __device__ inline v16bf fragA(P tile, int stride, int k0, int lane) {
    const int m   = lane & 15;
    const int sel = (lane >> 4) & 1;
    P p = tile + m * stride + k0 + sel * 8;
    bf16x8 lo = *(const bf16x8*)p;
    bf16x8 hi = *(const bf16x8*)(p + 16);
    return __builtin_shufflevector(lo, hi, 0,1,2,3,4,5,6,7,8,9,10,11,12,13,14,15);
}

// B-operand fragment (32x16 bf16, K x N) from a row-major [N x K] tile
// (i.e. computing A*W^T): lane = column n, same K order as fragA:
// low lanes K = {k0..k0+7, k0+16..k0+23}, high lanes the +8 halves.
template <typename P>
static __device__ inline v16bf fragB(P tile, int stride, int k0, int lane) {
    const int n   = lane & 15;
    const int sel = (lane >> 4) & 1;
    P p = tile + n * stride + k0 + sel * 8;
    bf16x8 lo = *(const bf16x8*)p;
    bf16x8 hi = *(const bf16x8*)(p + 16);
    return __builtin_shufflevector(lo, hi, 0,1,2,3,4,5,6,7,8,9,10,11,12,13,14,15);
}

static __device__ inline v8f wmma_bf16(v16bf a, v16bf b, v8f c) {
    return __builtin_amdgcn_wmma_f32_16x16x32_bf16(false, a, false, b, (short)0, c,
                                                   false, false);
}

// ---------------- weight fp32 -> bf16 pre-pass ----------------
__global__ void __launch_bounds__(256)
xattn_convert_weights(const float* __restrict__ Wq, const float* __restrict__ Wk,
                      const float* __restrict__ Wv, const float* __restrict__ Wo,
                      bf16* __restrict__ dst) {
    const long idx = ((long)blockIdx.x * blockDim.x + threadIdx.x) * 4;
    const long M = (long)DMODEL * DMODEL; // 1M elements per matrix
    if (idx >= 4 * M) return;
    const int which = (int)(idx >> 20);
    const long off  = idx & (M - 1);
    const float* src = (which == 0) ? Wq : (which == 1) ? Wk : (which == 2) ? Wv : Wo;
    float4 v = *(const float4*)(src + off);
    dst[idx + 0] = (bf16)v.x;
    dst[idx + 1] = (bf16)v.y;
    dst[idx + 2] = (bf16)v.z;
    dst[idx + 3] = (bf16)v.w;
}

// ---------------- fused cross-attention kernel ----------------
// LDS layout (bf16 elements), 320 KB total:
//   Qs[32*1024] Ks[32*1024] Vs[32*1024] As[32*1024] Cs[32*1024]
//   Fs (feats, 32*1024) aliases As; Ps (probs, 32*256) aliases Cs
//   (As/Cs are dead after the QKV GEMMs).
__global__ void __launch_bounds__(NTHREADS)
xattn_fused(const float* __restrict__ latent, const float* __restrict__ cond,
            const bf16* __restrict__ Wq, const bf16* __restrict__ Wk,
            const bf16* __restrict__ Wv, const bf16* __restrict__ Wo,
            const float* __restrict__ bout, float* __restrict__ out) {
    extern __shared__ __align__(16) char smem_raw[];
    bf16* Qs = (bf16*)smem_raw;
    bf16* Ks = Qs + ROWS * DMODEL;
    bf16* Vs = Ks + ROWS * DMODEL;
    bf16* As = Vs + ROWS * DMODEL;
    bf16* Cs = As + ROWS * DMODEL;
    bf16* Fs = As; // alias: feats after A consumed
    bf16* Ps = Cs; // alias: probs after C consumed

    const int tid  = threadIdx.x;
    const int lane = tid & 31;
    const int wid  = tid >> 5;
    const int n    = lane & 15;
    const int sel  = (lane >> 4) & 1;
    const long rowBase = (long)blockIdx.x * ROWS;

    // ---- stage latent/cond rows into LDS as bf16 ----
    const float* Ablk = latent + rowBase * DMODEL;
    const float* Cblk = cond   + rowBase * DMODEL;
    for (int i = tid; i < ROWS * DMODEL / 4; i += NTHREADS) {
        float4 a = ((const float4*)Ablk)[i];
        float4 c = ((const float4*)Cblk)[i];
        const int b4 = i * 4;
        As[b4 + 0] = (bf16)a.x; As[b4 + 1] = (bf16)a.y;
        As[b4 + 2] = (bf16)a.z; As[b4 + 3] = (bf16)a.w;
        Cs[b4 + 0] = (bf16)c.x; Cs[b4 + 1] = (bf16)c.y;
        Cs[b4 + 2] = (bf16)c.z; Cs[b4 + 3] = (bf16)c.w;
    }
    __syncthreads();

    // ---- GEMM: Dst[32,1024](LDS,bf16) = Asrc[32,1024] * W^T ----
    // Each weight fragment feeds ROWTILES WMMAs (L2-bandwidth amortization).
    auto gemm_lds = [&](const bf16* Asrc, const bf16* W, bf16* Dst) {
        v8f acc[ROWTILES][TILES_PER_WAVE];
#pragma unroll
        for (int rt = 0; rt < ROWTILES; ++rt)
#pragma unroll
            for (int i = 0; i < TILES_PER_WAVE; ++i) acc[rt][i] = zerof8();
        for (int k0 = 0; k0 < DMODEL; k0 += 32) {
            v16bf a0 = fragA(Asrc,               DMODEL, k0, lane);
            v16bf a1 = fragA(Asrc + 16 * DMODEL, DMODEL, k0, lane);
#pragma unroll
            for (int i = 0; i < TILES_PER_WAVE; ++i) {
                const int t = wid * TILES_PER_WAVE + i;
                v16bf w = fragB(W + (long)t * 16 * DMODEL, DMODEL, k0, lane);
                acc[0][i] = wmma_bf16(a0, w, acc[0][i]);
                acc[1][i] = wmma_bf16(a1, w, acc[1][i]);
            }
        }
#pragma unroll
        for (int rt = 0; rt < ROWTILES; ++rt)
#pragma unroll
            for (int i = 0; i < TILES_PER_WAVE; ++i) {
                const int t = wid * TILES_PER_WAVE + i;
#pragma unroll
                for (int r = 0; r < 8; ++r)
                    Dst[(rt * 16 + r + 8 * sel) * DMODEL + t * 16 + n] =
                        (bf16)acc[rt][i][r];
            }
    };

    gemm_lds(As, Wq, Qs);   // Q = latent @ Wq^T
    gemm_lds(Cs, Wk, Ks);   // K = cond   @ Wk^T
    gemm_lds(Cs, Wv, Vs);   // V = cond   @ Wv^T
    __syncthreads();

    // ---- attention across heads, ROWS_PER_WAVE batch rows per wave ----
    for (int b = wid * ROWS_PER_WAVE; b < (wid + 1) * ROWS_PER_WAVE; ++b) {
        const bf16* Qr = Qs + b * DMODEL; // [16 heads][64]
        const bf16* Kr = Ks + b * DMODEL;
        const bf16* Vr = Vs + b * DMODEL;
        bf16* Pr = Ps + b * 256;          // [16 h][16 g]

        // scores[h,g] = sum_d Q[h,d] K[g,d]   (16x16, K=64 via 2 WMMAs)
        v8f s = zerof8();
        s = wmma_bf16(fragA(Qr, DIM_HEAD, 0, lane),  fragB(Kr, DIM_HEAD, 0, lane),  s);
        s = wmma_bf16(fragA(Qr, DIM_HEAD, 32, lane), fragB(Kr, DIM_HEAD, 32, lane), s);

        // softmax over g (16 lanes of a half-wave group) per row h
#pragma unroll
        for (int r = 0; r < 8; ++r) {
            float v = s[r] * SCALE;
            float m = v;
            m = fmaxf(m, __shfl_xor(m, 1, 32));
            m = fmaxf(m, __shfl_xor(m, 2, 32));
            m = fmaxf(m, __shfl_xor(m, 4, 32));
            m = fmaxf(m, __shfl_xor(m, 8, 32));
            float e = __expf(v - m);
            float sum = e;
            sum += __shfl_xor(sum, 1, 32);
            sum += __shfl_xor(sum, 2, 32);
            sum += __shfl_xor(sum, 4, 32);
            sum += __shfl_xor(sum, 8, 32);
            Pr[(r + 8 * sel) * 16 + n] = (bf16)(e / sum);
        }

        // feats[h,d] = sum_g probs[h,g] V[g,d]  (16x64, K=16 zero-padded to 32)
        v16bf aP;
        {
            const bf16* p = Pr + (lane & 15) * 16 + sel * 8;
            bf16x8 lo = *(const bf16x8*)p;
            aP = __builtin_shufflevector(lo, zero8(),
                                         0,1,2,3,4,5,6,7,8,9,10,11,12,13,14,15);
        }
#pragma unroll
        for (int ct = 0; ct < 4; ++ct) {
            v16bf bv = zero16();
            {
#pragma unroll
                for (int j = 0; j < 8; ++j)
                    bv[j] = Vr[(j + 8 * sel) * DIM_HEAD + ct * 16 + n];
            }
            v8f f = wmma_bf16(aP, bv, zerof8());
#pragma unroll
            for (int r = 0; r < 8; ++r)
                Fs[b * DMODEL + (r + 8 * sel) * DIM_HEAD + ct * 16 + n] = (bf16)f[r];
        }
    }
    __syncthreads();

    // ---- output projection: out[32,1024] = F @ Wout^T + bout ----
    {
        v8f acc[ROWTILES][TILES_PER_WAVE];
#pragma unroll
        for (int rt = 0; rt < ROWTILES; ++rt)
#pragma unroll
            for (int i = 0; i < TILES_PER_WAVE; ++i) acc[rt][i] = zerof8();
        for (int k0 = 0; k0 < DMODEL; k0 += 32) {
            v16bf a0 = fragA(Fs,               DMODEL, k0, lane);
            v16bf a1 = fragA(Fs + 16 * DMODEL, DMODEL, k0, lane);
#pragma unroll
            for (int i = 0; i < TILES_PER_WAVE; ++i) {
                const int t = wid * TILES_PER_WAVE + i;
                v16bf w = fragB(Wo + (long)t * 16 * DMODEL, DMODEL, k0, lane);
                acc[0][i] = wmma_bf16(a0, w, acc[0][i]);
                acc[1][i] = wmma_bf16(a1, w, acc[1][i]);
            }
        }
#pragma unroll
        for (int rt = 0; rt < ROWTILES; ++rt)
#pragma unroll
            for (int i = 0; i < TILES_PER_WAVE; ++i) {
                const int t = wid * TILES_PER_WAVE + i;
                const int col = t * 16 + n;
                const float bias = bout[col];
#pragma unroll
                for (int r = 0; r < 8; ++r)
                    out[(rowBase + rt * 16 + r + 8 * sel) * DMODEL + col] =
                        acc[rt][i][r] + bias;
            }
    }
}

// ---------------- launcher ----------------
extern "C" void kernel_launch(void* const* d_in, const int* in_sizes, int n_in,
                              void* d_out, int out_size, void* d_ws, size_t ws_size,
                              hipStream_t stream) {
    const float* latent = (const float*)d_in[0];
    const float* cond   = (const float*)d_in[1];
    const float* Wq     = (const float*)d_in[2];
    const float* Wk     = (const float*)d_in[3];
    const float* Wv     = (const float*)d_in[4];
    const float* Wo     = (const float*)d_in[5];
    const float* bout   = (const float*)d_in[6];
    float* out = (float*)d_out;

    bf16* wbf = (bf16*)d_ws; // 4 * 1M bf16 = 8 MB
    const long M = (long)DMODEL * DMODEL;

    // weights fp32 -> bf16 (small; stays hot in the 192MB L2 afterwards)
    xattn_convert_weights<<<(4 * M) / (256 * 4), 256, 0, stream>>>(Wq, Wk, Wv, Wo, wbf);

    const size_t ldsBytes = (size_t)5 * ROWS * DMODEL * sizeof(bf16); // 320 KB
    (void)hipFuncSetAttribute((const void*)xattn_fused,
                              hipFuncAttributeMaxDynamicSharedMemorySize,
                              (int)ldsBytes);
    xattn_fused<<<BATCH / ROWS, NTHREADS, ldsBytes, stream>>>(
        latent, cond, wbf, wbf + M, wbf + 2 * M, wbf + 3 * M, bout, out);
}
